// MultiHeadAttention_84327387889772
// MI455X (gfx1250) — compile-verified, matrix operand order checked by reading, not run on hardware
//
#include <hip/hip_runtime.h>

// ---------------------------------------------------------------------------
// MI455X (gfx1250, wave32) fused multi-head attention.
// Roofline: the 512MB fp32 attn output dominates HBM traffic (~22us floor at
// 23.3 TB/s); total FLOPs (~70GF) are negligible against the bf16 WMMA path,
// so everything is organized to (a) write attn exactly once (flash-style
// recompute of QK^T instead of spilling logits), (b) keep K/V bf16 and
// L2-resident (8MB/head-tensor vs 192MB L2), (c) run all GEMMs through
// v_wmma_f32_16x16x32_bf16 with register-pipelined LDS staging.
// ---------------------------------------------------------------------------

#define D_MODEL 1024
#define NUM_HEADS 16
#define DEPTH 64
#define SEQ 2048
#define BATCH 2
#define NEG_BIG (-1.0e9f)

typedef __attribute__((ext_vector_type(16))) __bf16 v16bf;
typedef __attribute__((ext_vector_type(8)))  float  v8f;

struct FragBits { unsigned int d[8]; };

// Gather one 16x32 bf16 WMMA operand (A layout; B uses the same pattern with
// "line" = N column) from a row-major [16][stride] bf16 tile (global or LDS).
// ISA 7.12.2: lanes 0-15 hold line=lane with K {0..7,16..23}; lanes 16-31 hold
// the same line with K {8..15,24..31}; each VGPR packs 2 consecutive K halves,
// so each half-fragment is a contiguous 16B run -> ds_load_b128 /
// global_load_b128 after compiler merging.
__device__ inline v16bf load_frag_16x32(const __bf16* base, int stride, int lane) {
  const int line = lane & 15;
  const int kb   = (lane & 16) ? 8 : 0;
  const __bf16* rp = base + (size_t)line * stride;
  FragBits f;
#pragma unroll
  for (int j = 0; j < 4; ++j)
    f.d[j] = *reinterpret_cast<const unsigned int*>(rp + kb + 2 * j);
#pragma unroll
  for (int j = 0; j < 4; ++j)
    f.d[4 + j] = *reinterpret_cast<const unsigned int*>(rp + 16 + kb + 2 * j);
  return __builtin_bit_cast(v16bf, f);
}

__device__ inline v8f wmma_bf16(v16bf a, v16bf b, v8f c) {
  return __builtin_amdgcn_wmma_f32_16x16x32_bf16(
      false, a, false, b, (short)0, c, false, false);
}

__device__ inline unsigned int pack_bf16x2(float a, float b) {
  union { __bf16 h[2]; unsigned int u; } p;
  p.h[0] = (__bf16)a;
  p.h[1] = (__bf16)b;
  return p.u;           // lowers to v_cvt_pk_bf16_f32
}

// ---------------------------------------------------------------------------
// Generic tiled GEMM: Y = X(Mx1024) @ W(1024x1024) + bias, M = 4096.
// 64x64 tile per 128-thread block (4 waves; each wave owns a 16x64 strip).
// K-loop is register-pipelined: chunk k+1 is fetched with global_load_b128
// while chunk k's WMMAs consume LDS, so loads hide behind matrix math.
// mode 0: store bf16 split-heads (B,H,S,64)             (Q, K projections)
// mode 1: store bf16 split-heads transposed (B,H,64,S)  (V projection)
// mode 2: store fp32 row-major (M, 1024)                (output projection)
// ---------------------------------------------------------------------------
__global__ __launch_bounds__(128)
void mha_gemm_kernel(const float* __restrict__ X, const float* __restrict__ W,
                     const float* __restrict__ bias, void* __restrict__ out,
                     int mode) {
  __shared__ __bf16 Xs[64][32];   // [m][k]
  __shared__ __bf16 Ws[64][32];   // [n][k] (transposed stage -> contiguous K)

  const int tid  = threadIdx.x;
  const int wave = tid >> 5;
  const int lane = tid & 31;
  const int m0 = blockIdx.x * 64;
  const int n0 = blockIdx.y * 64;

  // Per-thread staging registers: X tile 64x32 = 512 float4 (4/thread),
  // W tile 32x64 = 512 float4 (4/thread).
  const int xr = (tid * 4) >> 5, xq = (tid * 4 >> 2) & 7;   // idx = i*512+tid*4
  const int wk = (tid * 4) >> 6, wq = tid & 15;

  float4 xv[4], wv[4];
#pragma unroll
  for (int i = 0; i < 4; ++i) {           // preload chunk 0 (batched b128 loads)
    const int idx = i * 512 + tid * 4;    // 0..2047 in float units, 4-aligned
    const int r = idx >> 5, kq = (idx & 31) >> 2;
    xv[i] = ((const float4*)(X + (size_t)(m0 + r) * D_MODEL))[kq];
  }
#pragma unroll
  for (int i = 0; i < 4; ++i) {
    const int idx = i * 512 + tid * 4;
    const int k = idx >> 6, nq = (idx & 63) >> 2;
    wv[i] = ((const float4*)(W + (size_t)k * D_MODEL + n0))[nq];
  }

  v8f acc[4] = {};

  for (int kc = 0; kc < D_MODEL; kc += 32) {
    // Commit staged registers to LDS as bf16 (pk-convert, 8B stores for X).
#pragma unroll
    for (int i = 0; i < 4; ++i) {
      const int idx = i * 512 + tid * 4;
      const int r = idx >> 5, kq = (idx & 31) >> 2;
      uint2 pk;
      pk.x = pack_bf16x2(xv[i].x, xv[i].y);
      pk.y = pack_bf16x2(xv[i].z, xv[i].w);
      *reinterpret_cast<uint2*>(&Xs[r][kq * 4]) = pk;
    }
#pragma unroll
    for (int i = 0; i < 4; ++i) {
      const int idx = i * 512 + tid * 4;
      const int k = idx >> 6, nq = (idx & 63) >> 2;
      Ws[nq * 4 + 0][k] = (__bf16)wv[i].x;   // transpose -> contiguous K rows
      Ws[nq * 4 + 1][k] = (__bf16)wv[i].y;
      Ws[nq * 4 + 2][k] = (__bf16)wv[i].z;
      Ws[nq * 4 + 3][k] = (__bf16)wv[i].w;
    }
    __syncthreads();

    // Kick off next chunk's global loads; they overlap the WMMAs below.
    const int kn = kc + 32;
    if (kn < D_MODEL) {
#pragma unroll
      for (int i = 0; i < 4; ++i) {
        const int idx = i * 512 + tid * 4;
        const int r = idx >> 5, kq = (idx & 31) >> 2;
        xv[i] = ((const float4*)(X + (size_t)(m0 + r) * D_MODEL + kn))[kq];
      }
#pragma unroll
      for (int i = 0; i < 4; ++i) {
        const int idx = i * 512 + tid * 4;
        const int k = idx >> 6, nq = (idx & 63) >> 2;
        wv[i] = ((const float4*)(W + (size_t)(kn + k) * D_MODEL + n0))[nq];
      }
    }

    v16bf a = load_frag_16x32(&Xs[wave * 16][0], 32, lane);
#pragma unroll
    for (int nt = 0; nt < 4; ++nt) {
      v16bf b = load_frag_16x32(&Ws[nt * 16][0], 32, lane);
      acc[nt] = wmma_bf16(a, b, acc[nt]);
    }
    __syncthreads();
  }

  // Epilogue. C/D layout: lane -> column (lane&15), VGPR r -> row
  // (r for lanes 0-15, r+8 for lanes 16-31).
  const int ncol  = lane & 15;
  const int rbase = (lane & 16) ? 8 : 0;
#pragma unroll
  for (int nt = 0; nt < 4; ++nt) {
    const int c  = n0 + nt * 16 + ncol;
    const float bvv = bias[c];
#pragma unroll
    for (int r = 0; r < 8; ++r) {
      const int m = m0 + wave * 16 + rbase + r;
      const float val = acc[nt][r] + bvv;
      if (mode == 2) {
        ((float*)out)[(size_t)m * D_MODEL + c] = val;
      } else {
        const int b = m >> 11, s = m & (SEQ - 1);
        const int h = c >> 6,  d = c & (DEPTH - 1);
        __bf16* o = (__bf16*)out;
        if (mode == 0)
          o[(((size_t)(b * NUM_HEADS + h)) * SEQ + s) * DEPTH + d] = (__bf16)val;
        else
          o[(((size_t)(b * NUM_HEADS + h)) * DEPTH + d) * SEQ + s] = (__bf16)val;
      }
    }
  }
}

// ---------------------------------------------------------------------------
// Fused attention: one wave32 per (b, h, 16 query rows).
// Pass 1: QK^T via WMMA, online row max / exp-sum only (flash statistics).
// Pass 2: recompute QK^T (compute is free; avoids spilling logits), write the
// normalized attn fp32 exactly once, transpose probs through a 1KB LDS tile,
// and accumulate ctx = P @ V with WMMA against pre-transposed V (B,H,64,S).
// ---------------------------------------------------------------------------
__global__ __launch_bounds__(32)
void mha_attn_kernel(const __bf16* __restrict__ qh, const __bf16* __restrict__ kh,
                     const __bf16* __restrict__ vhT, const float* __restrict__ mask,
                     float* __restrict__ attn_out, float* __restrict__ ctx) {
  __shared__ __bf16 pstage[16][32];   // probability transpose staging (C -> A layout)

  const int lane = threadIdx.x & 31;
  const int blk  = blockIdx.x;        // (b*H + h) * 128 + qblock
  const int qb   = blk & 127;
  const int bh   = blk >> 7;
  const int b    = bh >> 4;
  const int h    = bh & 15;

  const __bf16* qbase = qh  + ((size_t)bh * SEQ + qb * 16) * DEPTH;
  const __bf16* kbase = kh  + (size_t)bh * SEQ * DEPTH;
  const __bf16* vbase = vhT + (size_t)bh * DEPTH * SEQ;
  const float*  mrow  = mask + (size_t)b * SEQ;

  // Q fragments: 16x64 = two 16x32 A fragments (K = depth).
  const v16bf a0 = load_frag_16x32(qbase,      DEPTH, lane);
  const v16bf a1 = load_frag_16x32(qbase + 32, DEPTH, lane);

  const float scale = 0.125f;               // 1/sqrt(DEPTH)
  const int ncol  = lane & 15;              // C-layout column owned by lane
  const int rbase = (lane & 16) ? 8 : 0;    // C-layout row base

  float m[8], l[8];
#pragma unroll
  for (int r = 0; r < 8; ++r) { m[r] = -3.0e38f; l[r] = 0.0f; }

  // ---------------- pass 1: online softmax statistics ----------------
  for (int kt = 0; kt < SEQ / 16; ++kt) {
    const __bf16* kp = kbase + (size_t)kt * 16 * DEPTH;
    if (kt + 1 < SEQ / 16)
      __builtin_prefetch(kp + 16 * DEPTH + lane * 32, 0, 0);
    const v16bf b0 = load_frag_16x32(kp,      DEPTH, lane);
    const v16bf b1 = load_frag_16x32(kp + 32, DEPTH, lane);
    v8f c = {};
    c = wmma_bf16(a0, b0, c);
    c = wmma_bf16(a1, b1, c);
    const float mv = mrow[kt * 16 + ncol] * NEG_BIG;
#pragma unroll
    for (int r = 0; r < 8; ++r) {
      float x = c[r] * scale + mv;
      float tm = x;                               // row max across 16 lanes
      tm = fmaxf(tm, __shfl_xor(tm, 1));
      tm = fmaxf(tm, __shfl_xor(tm, 2));
      tm = fmaxf(tm, __shfl_xor(tm, 4));
      tm = fmaxf(tm, __shfl_xor(tm, 8));
      const float mn = fmaxf(m[r], tm);
      float p = __expf(x - mn);
      float ps = p;                               // row sum across 16 lanes
      ps += __shfl_xor(ps, 1);
      ps += __shfl_xor(ps, 2);
      ps += __shfl_xor(ps, 4);
      ps += __shfl_xor(ps, 8);
      l[r] = l[r] * __expf(m[r] - mn) + ps;
      m[r] = mn;
    }
  }
  float inv_l[8];
#pragma unroll
  for (int r = 0; r < 8; ++r) inv_l[r] = 1.0f / l[r];

  // ---------------- pass 2: recompute, emit attn, P @ V ----------------
  v8f acc[4] = {};
  float* arow = attn_out + ((size_t)bh * SEQ + qb * 16) * SEQ;

  for (int kt2 = 0; kt2 < SEQ / 32; ++kt2) {
#pragma unroll
    for (int sub = 0; sub < 2; ++sub) {
      const int kt = kt2 * 2 + sub;
      const __bf16* kp = kbase + (size_t)kt * 16 * DEPTH;
      const v16bf b0 = load_frag_16x32(kp,      DEPTH, lane);
      const v16bf b1 = load_frag_16x32(kp + 32, DEPTH, lane);
      v8f c = {};
      c = wmma_bf16(a0, b0, c);
      c = wmma_bf16(a1, b1, c);
      const float mv = mrow[kt * 16 + ncol] * NEG_BIG;
#pragma unroll
      for (int r = 0; r < 8; ++r) {
        const float x = c[r] * scale + mv;
        const float p = __expf(x - m[r]) * inv_l[r];
        const int row = rbase + r;
        arow[(size_t)row * SEQ + kt * 16 + ncol] = p;    // the one attn write
        pstage[row][sub * 16 + ncol] = (__bf16)p;        // C -> A transpose
      }
    }
    __syncthreads();   // single-wave WG: ds-count fence only

    const v16bf pa = load_frag_16x32(&pstage[0][0], 32, lane);
#pragma unroll
    for (int nt = 0; nt < 4; ++nt) {
      // B fragment of V: K = 32 keys (contiguous in vhT rows), N = 16 depths.
      const v16bf bv =
          load_frag_16x32(vbase + (size_t)(nt * 16) * SEQ + kt2 * 32, SEQ, lane);
      acc[nt] = wmma_bf16(pa, bv, acc[nt]);
    }
    __syncthreads();
  }

  // ctx written merged-heads (B, S, D_MODEL) fp32, ready for the Wo GEMM.
#pragma unroll
  for (int nt = 0; nt < 4; ++nt) {
#pragma unroll
    for (int r = 0; r < 8; ++r) {
      const int s = qb * 16 + rbase + r;
      ctx[((size_t)b * SEQ + s) * D_MODEL + h * DEPTH + nt * 16 + ncol] = acc[nt][r];
    }
  }
}

// ---------------------------------------------------------------------------
extern "C" void kernel_launch(void* const* d_in, const int* in_sizes, int n_in,
                              void* d_out, int out_size, void* d_ws, size_t ws_size,
                              hipStream_t stream) {
  const float* v    = (const float*)d_in[0];
  const float* k    = (const float*)d_in[1];
  const float* q    = (const float*)d_in[2];
  const float* mask = (const float*)d_in[3];
  const float* Wq   = (const float*)d_in[4];
  const float* bq   = (const float*)d_in[5];
  const float* Wk   = (const float*)d_in[6];
  const float* bk   = (const float*)d_in[7];
  const float* Wv   = (const float*)d_in[8];
  const float* bv   = (const float*)d_in[9];
  const float* Wo   = (const float*)d_in[10];
  const float* bo   = (const float*)d_in[11];

  // Workspace: qh/kh/vhT bf16 (8MB each) + ctx fp32 (16MB) = 40MB.
  char* ws = (char*)d_ws;
  __bf16* qh  = (__bf16*)(ws);
  __bf16* kh  = (__bf16*)(ws + (size_t)(8ull << 20));
  __bf16* vhT = (__bf16*)(ws + (size_t)(16ull << 20));
  float*  ctx = (float*) (ws + (size_t)(24ull << 20));

  float* out  = (float*)d_out;                               // (B,S,D) fp32
  float* attn = out + (size_t)BATCH * SEQ * D_MODEL;         // (B,H,S,S) fp32

  const dim3 gg(BATCH * SEQ / 64, D_MODEL / 64);
  const dim3 gb(128);

  mha_gemm_kernel<<<gg, gb, 0, stream>>>(q, Wq, bq, (void*)qh,  0);
  mha_gemm_kernel<<<gg, gb, 0, stream>>>(k, Wk, bk, (void*)kh,  0);
  mha_gemm_kernel<<<gg, gb, 0, stream>>>(v, Wv, bv, (void*)vhT, 1);

  mha_attn_kernel<<<BATCH * NUM_HEADS * (SEQ / 16), 32, 0, stream>>>(
      qh, kh, vhT, mask, attn, ctx);

  mha_gemm_kernel<<<gg, gb, 0, stream>>>(ctx, Wo, bo, (void*)out, 2);
}
